// StageNet_I_36472862278257
// MI455X (gfx1250) — matrix operand order repaired and checked by reading, not compiled, not run on hardware
//
#include <hip/hip_runtime.h>
#include <hip/hip_bf16.h>

// StageNet recurrence on MI455X (gfx1250), wave32 + WMMA f16->f32.
//   B=256 T=400 F=76 H=384 L=3 K=10 O=1, G=4H+2L (rows 0..5 of z unused)
// Per step: z-GEMM (256x1536x461) + conv-GEMM (256x384x3840) via
// v_wmma_f32_16x16x32_f16; ~458 GFLOP total, everything L2-resident.

typedef __attribute__((ext_vector_type(16))) _Float16 v16h;
typedef __attribute__((ext_vector_type(8)))  _Float16 v8h;
typedef __attribute__((ext_vector_type(8)))  float    v8f;

#define Bn  256
#define Tn  400
#define Fn  76
#define Hn  384
#define Kn  10
#define GN  1536   // 4*H (used part of z)
#define KA  480    // padded K-dim of z-GEMM (76 + 384 + 1 -> 480 = 15*32)
#define SCn 64

// workspace layout (bytes, all 32B+ aligned)
#define WCAT_OFF 0u         // f16 [1536][480]
#define BSUM_OFF 1474560u   // f32 [1536]
#define WC16_OFF 1480704u   // f16 [10][384][384]  (k-major repack of Wc)
#define C_OFF    4429824u   // f32 [256][384]  cell state
#define HSUM_OFF 4823040u   // f32 [256][384]  running sum of ring (for mean)
#define HBUF_OFF 5216256u   // f16 [10][256][384]  h ring buffer
#define CONV_OFF 7182336u   // f32 [256][384]  conv scratch

__device__ __forceinline__ float sigmoidf_(float x){ return 1.0f/(1.0f+__expf(-x)); }

// ---------------- one-time weight repack + state zero ----------------
__global__ void prep_kernel(const float* __restrict__ Wk, const float* __restrict__ bk,
                            const float* __restrict__ Wr, const float* __restrict__ br,
                            const float* __restrict__ Wc,
                            _Float16* __restrict__ Wcat, float* __restrict__ bsum,
                            _Float16* __restrict__ Wc16, float* __restrict__ c,
                            float* __restrict__ hsum, _Float16* __restrict__ hbuf)
{
  size_t idx    = (size_t)blockIdx.x*blockDim.x + threadIdx.x;
  size_t stride = (size_t)gridDim.x*blockDim.x;
  // Wcat[n][col]: col<76 -> Wk[6+n][col]; 76..459 -> Wr[6+n][col-76];
  // 460 -> Wk[6+n][76]+Wr[6+n][384] (time column); 461..479 -> 0 pad
  for (size_t i = idx; i < (size_t)GN*KA; i += stride) {
    int n = (int)(i/KA), col = (int)(i%KA);
    float v;
    if      (col < Fn)      v = Wk[(size_t)(6+n)*(Fn+1) + col];
    else if (col < Fn+Hn)   v = Wr[(size_t)(6+n)*(Hn+1) + (col-Fn)];
    else if (col == Fn+Hn)  v = Wk[(size_t)(6+n)*(Fn+1)+Fn] + Wr[(size_t)(6+n)*(Hn+1)+Hn];
    else                    v = 0.0f;
    Wcat[i] = (_Float16)v;
  }
  for (size_t i = idx; i < (size_t)GN; i += stride) bsum[i] = bk[6+i] + br[6+i];
  // Wc16[k][o][i] = Wc[o][i][k]  (k-major: each ring slot is one dense GEMM)
  for (size_t i = idx; i < (size_t)Kn*Hn*Hn; i += stride) {
    int k  = (int)(i/(Hn*Hn));
    int r  = (int)(i%(Hn*Hn));
    int o  = r/Hn, ii = r%Hn;
    Wc16[i] = (_Float16)Wc[((size_t)o*Hn + ii)*Kn + k];
  }
  for (size_t i = idx; i < (size_t)Bn*Hn; i += stride) { c[i]=0.f; hsum[i]=0.f; }
  for (size_t i = idx; i < (size_t)Kn*Bn*Hn; i += stride) hbuf[i] = (_Float16)0.f;
}

// ---------------- per-step: z GEMM + LSTM cell update -----------------
// grid (16, 24): M-tile(batch) x P-tile(channel p in [0,384)).
// 4 waves: wave w computes z-tile at n = w*384 + P0 (the f/i/o/g blocks
// for the same 16 channels) -> fused gate math in one block.
__global__ __launch_bounds__(128)
void z_gate_kernel(const float* __restrict__ x, const float* __restrict__ tvec,
                   const _Float16* __restrict__ Wcat, const float* __restrict__ bsum,
                   float* __restrict__ c, float* __restrict__ hsum,
                   _Float16* __restrict__ hbuf, int t)
{
  __shared__ __align__(32) _Float16 At[16][KA+8];   // A tile: 16 rows x 480 (+pad)
  __shared__ float zb[4][16][16];                   // z tiles per gate block
  const int tid = threadIdx.x;
  const int M0 = blockIdx.x*16;
  const int P0 = blockIdx.y*16;
  const int slot_prev = (t + Kn - 1) % Kn;          // ring slot of h_{t-1}

  // Build A = [x_t | h_{t-1} | time | 0pad] in f16
  for (int idx = tid; idx < 16*KA; idx += 128) {
    int r = idx/KA, col = idx%KA;
    int b = M0 + r;
    float v;
    if      (col < Fn)      v = x[((size_t)b*Tn + t)*Fn + col];
    else if (col < Fn+Hn)   v = (float)hbuf[((size_t)slot_prev*Bn + b)*Hn + (col-Fn)];
    else if (col == Fn+Hn)  v = tvec[(size_t)b*Tn + t];
    else                    v = 0.0f;
    At[r][col] = (_Float16)v;
  }
  __syncthreads();

  const int wave = tid>>5, lane = tid&31;
  const int n0   = wave*Hn + P0;          // z column base for this wave
  const int rowA = lane&15;
  const int hiA  = (lane>>4)*16;          // A: K 0..15 / 16..31 per lane half
  const int nB   = n0 + (lane&15);
  const int kbB  = (lane>>4)*16;          // B: K 0..15 / 16..31 per lane half
  v8f acc = {};
  #pragma unroll
  for (int k = 0; k < KA/32; ++k) {
    v8h alo = *(const v8h*)&At[rowA][k*32 + hiA];
    v8h ahi = *(const v8h*)&At[rowA][k*32 + hiA + 8];
    v16h a  = __builtin_shufflevector(alo, ahi, 0,1,2,3,4,5,6,7,8,9,10,11,12,13,14,15);
    v16h bf = *(const v16h*)(Wcat + (size_t)nB*KA + k*32 + kbB);
    acc = __builtin_amdgcn_wmma_f32_16x16x32_f16(false, a, false, bf,
                                                 (short)0, acc, false, false);
  }
  { // spill z tile (+bias) to LDS; D layout: lane<16 -> M=r, lane>=16 -> M=8+r
    const int nl = lane&15;
    const float bias = bsum[n0 + nl];
    const int mh = (lane>>4)*8;
    #pragma unroll
    for (int r = 0; r < 8; ++r) zb[wave][mh+r][nl] = acc[r] + bias;
  }
  __syncthreads();

  // LSTM cell update for the 16x16 (batch, channel) patch
  const int slot_new = t % Kn;
  for (int q = tid; q < 256; q += 128) {
    int m = q>>4, pl = q&15;
    int b = M0 + m, p = P0 + pl;
    float zf = zb[0][m][pl], zi = zb[1][m][pl];
    float zo = zb[2][m][pl], zg = zb[3][m][pl];
    size_t cidx = (size_t)b*Hn + p;
    float cn = sigmoidf_(zf)*c[cidx] + sigmoidf_(zi)*tanhf(zg);
    float hn = sigmoidf_(zo)*tanhf(cn);
    c[cidx] = cn;
    size_t hidx = ((size_t)slot_new*Bn + b)*Hn + p;
    float hold = (float)hbuf[hidx];       // h_{t-10} being evicted
    hsum[cidx] += hn - hold;              // keep running sum for theme mean
    hbuf[hidx] = (_Float16)hn;
  }
}

// ---------------- per-step: temporal conv GEMM (K = 10*384) -----------
// conv[b,o] = sum_k sum_i ring[(t+1+k)%10][b][i] * Wc16[k][o][i]
__global__ __launch_bounds__(32)
void conv_kernel(const _Float16* __restrict__ hbuf, const _Float16* __restrict__ Wc16,
                 const float* __restrict__ bc, float* __restrict__ convw, int t)
{
  const int lane = threadIdx.x & 31;
  const int M0 = blockIdx.x*16;           // batch tile
  const int N0 = blockIdx.y*16;           // output-channel tile
  const int rowA = M0 + (lane&15);
  const int hiA  = (lane>>4)*16;
  const int nB   = N0 + (lane&15);
  const int kbB  = (lane>>4)*16;
  v8f acc = {};
  for (int k = 0; k < Kn; ++k) {
    const int slot = (t + 1 + k) % Kn;    // buf[k] = h_{t-9+k}
    const _Float16* Ab = hbuf + ((size_t)slot*Bn + rowA)*Hn;
    const _Float16* Bb = Wc16 + ((size_t)k*Hn + nB)*Hn;
    #pragma unroll
    for (int kk = 0; kk < Hn/32; ++kk) {
      v8h alo = *(const v8h*)(Ab + kk*32 + hiA);
      v8h ahi = *(const v8h*)(Ab + kk*32 + hiA + 8);
      v16h a  = __builtin_shufflevector(alo, ahi, 0,1,2,3,4,5,6,7,8,9,10,11,12,13,14,15);
      v16h bf = *(const v16h*)(Bb + kk*32 + kbB);
      acc = __builtin_amdgcn_wmma_f32_16x16x32_f16(false, a, false, bf,
                                                   (short)0, acc, false, false);
    }
  }
  const int nl = lane&15;
  const float bias = bc[N0 + nl];
  const int mh = (lane>>4)*8;
  #pragma unroll
  for (int r = 0; r < 8; ++r)
    convw[(size_t)(M0+mh+r)*Hn + N0 + nl] = acc[r] + bias;
}

// ---------------- per-step: theme + output scalar ---------------------
// out[b,t] = sigmoid( sum_n (theme[b,n]*conv[b,n] + h_t[b,n]) * Wo[n] + bo )
__global__ __launch_bounds__(64)
void out_kernel(const float* __restrict__ Ws,  const float* __restrict__ bs,
                const float* __restrict__ Wrs, const float* __restrict__ brs,
                const float* __restrict__ Wo,  const float* __restrict__ bo,
                const float* __restrict__ hsum, const _Float16* __restrict__ hbuf,
                const float* __restrict__ convw, float* __restrict__ out, int t)
{
  __shared__ float hm[Hn];
  __shared__ float sv[SCn];
  __shared__ float red[64];
  const int b = blockIdx.x, tid = threadIdx.x;
  for (int i = tid; i < Hn; i += 64) hm[i] = hsum[(size_t)b*Hn + i]*0.1f;  // mean over K=10
  __syncthreads();
  {
    float acc = bs[tid];
    const float* wr = Ws + (size_t)tid*Hn;
    for (int i = 0; i < Hn; ++i) acc += wr[i]*hm[i];
    sv[tid] = fmaxf(acc, 0.0f);
  }
  __syncthreads();
  const int slot = t % Kn;                // newest h = h_t
  float partial = 0.0f;
  for (int n = tid; n < Hn; n += 64) {
    float acc = brs[n];
    const float* wr = Wrs + (size_t)n*SCn;
    #pragma unroll
    for (int j = 0; j < SCn; ++j) acc += wr[j]*sv[j];
    float theme = sigmoidf_(acc);
    float h = (float)hbuf[((size_t)slot*Bn + b)*Hn + n];
    partial += (theme*convw[(size_t)b*Hn + n] + h) * Wo[n];
  }
  red[tid] = partial;
  __syncthreads();
  if (tid == 0) {
    float s = 0.0f;
    for (int i = 0; i < 64; ++i) s += red[i];
    out[(size_t)b*Tn + t] = sigmoidf_(s + bo[0]);
  }
}

extern "C" void kernel_launch(void* const* d_in, const int* in_sizes, int n_in,
                              void* d_out, int out_size, void* d_ws, size_t ws_size,
                              hipStream_t stream)
{
  const float* x    = (const float*)d_in[0];
  const float* tvec = (const float*)d_in[1];
  const float* Wk   = (const float*)d_in[2];
  const float* bk   = (const float*)d_in[3];
  const float* Wr   = (const float*)d_in[4];
  const float* br   = (const float*)d_in[5];
  const float* Ws   = (const float*)d_in[6];
  const float* bs   = (const float*)d_in[7];
  const float* Wrs  = (const float*)d_in[8];
  const float* brs  = (const float*)d_in[9];
  const float* Wc   = (const float*)d_in[10];
  const float* bc   = (const float*)d_in[11];
  const float* Wo   = (const float*)d_in[12];
  const float* bo   = (const float*)d_in[13];

  char* ws = (char*)d_ws;
  _Float16* Wcat = (_Float16*)(ws + WCAT_OFF);
  float*    bsum = (float*)   (ws + BSUM_OFF);
  _Float16* Wc16 = (_Float16*)(ws + WC16_OFF);
  float*    c    = (float*)   (ws + C_OFF);
  float*    hsum = (float*)   (ws + HSUM_OFF);
  _Float16* hbuf = (_Float16*)(ws + HBUF_OFF);
  float*    convw= (float*)   (ws + CONV_OFF);
  float*    out  = (float*)d_out;

  prep_kernel<<<5760, 256, 0, stream>>>(Wk, bk, Wr, br, Wc,
                                        Wcat, bsum, Wc16, c, hsum, hbuf);
  for (int t = 0; t < Tn; ++t) {
    z_gate_kernel<<<dim3(16,24), 128, 0, stream>>>(x, tvec, Wcat, bsum, c, hsum, hbuf, t);
    conv_kernel  <<<dim3(16,24),  32, 0, stream>>>(hbuf, Wc16, bc, convw, t);
    out_kernel   <<<Bn,           64, 0, stream>>>(Ws, bs, Wrs, brs, Wo, bo,
                                                   hsum, hbuf, convw, out, t);
  }
}
